// GCN2_42331197669873
// MI455X (gfx1250) — compile-verified, matrix operand order checked by reading, not run on hardware
//
#include <hip/hip_runtime.h>
#include <hip/hip_bf16.h>
#include <math.h>

// ---------------------------------------------------------------------------
// GCN2 forward for MI455X (gfx1250, wave32, WMMA).
// Dense GEMMs: v_wmma_f32_16x16x32_f16, 32 rows x 128 cols per 256-thr block,
// B-fragment reuse across two row tiles. r2 staging uses the CDNA5
// global_load_async_to_lds_b128 path (ASYNCcnt).
// Sparse propagate: wave-per-edge global f32 atomics (agg fits in 192MB L2).
// ---------------------------------------------------------------------------

#define N_NODES 100000
#define E_EDGES 800000
#define NCLS    47
#define HDIM    128
#define NH      (N_NODES * HDIM)          // 12,800,000
#define NBLK32  (N_NODES / 32)            // 3125 row tiles of 32 (exact)

typedef __attribute__((ext_vector_type(16))) _Float16 v16h;
typedef __attribute__((ext_vector_type(8)))  _Float16 v8h;
typedef __attribute__((ext_vector_type(8)))  float    v8f;

// Load a 16-wide f16 WMMA fragment from an LDS row (A: row of M, B: column of N).
// ISA 7.12.2 16-bit A 16x32 layout: lanes 0-15 -> K {0..7, 16..23}+32*kk,
// lanes 16-31 -> K {8..15, 24..31}+32*kk. B symmetric (N on lanes).
__device__ __forceinline__ v16h load_frag(const _Float16* base, int kk, int half) {
    const _Float16* q = base + kk * 32 + half * 8;
    v8h lo = *(const v8h*)(q);
    v8h hi = *(const v8h*)(q + 16);
    v16h a;
#pragma unroll
    for (int j = 0; j < 8; ++j) { a[j] = lo[j]; a[j + 8] = hi[j]; }
    return a;
}

// ---------------- generic zero ----------------
__global__ void k_zero(float* __restrict__ p, long n) {
    long stride = (long)gridDim.x * blockDim.x;
    for (long i = (long)blockIdx.x * blockDim.x + threadIdx.x; i < n; i += stride)
        p[i] = 0.0f;
}

// ---------------- per-class counts (labels constant across layers) ----------
__global__ void k_count(const int* __restrict__ label, float* __restrict__ cnt, int n) {
    int i = blockIdx.x * blockDim.x + threadIdx.x;
    if (i < n) atomicAdd(&cnt[label[i]], 1.0f);
}

// ---------------- edge scatter: agg[dst] += w_e * x[src] --------------------
// one wave32 per edge, 4 floats per lane (row = 512B, fully coalesced)
__global__ void k_scatter(const int* __restrict__ src, const int* __restrict__ dst,
                          const float* __restrict__ ew, const float* __restrict__ X,
                          float* __restrict__ AGG) {
    int e    = blockIdx.x * 8 + (threadIdx.x >> 5);
    int lane = threadIdx.x & 31;
    int s = src[e], d = dst[e];
    float w = ew[e];
    const float4 xv = ((const float4*)(X + (long)s * HDIM))[lane];
    long base = (long)d * HDIM + lane * 4;
    atomicAdd(&AGG[base + 0], xv.x * w);
    atomicAdd(&AGG[base + 1], xv.y * w);
    atomicAdd(&AGG[base + 2], xv.z * w);
    atomicAdd(&AGG[base + 3], xv.w * w);
}

// ---------------- class sums: sums[label[n]] += x[n] + 0.1*x0[n] ------------
// LDS-staged partials (ds_add_f32), one flush of 6016 global atomics per block
__global__ void k_classsum(const float* __restrict__ X, const float* __restrict__ X0,
                           const int* __restrict__ label, float* __restrict__ sums) {
    __shared__ float sSum[NCLS * HDIM];
    int tid = threadIdx.x;
    for (int i = tid; i < NCLS * HDIM; i += 256) sSum[i] = 0.0f;
    __syncthreads();
    int h = tid & 127, sub = tid >> 7;                 // 2 nodes per block-step
    long step = (long)gridDim.x * 2;
    for (long n = (long)blockIdx.x * 2 + sub; n < N_NODES; n += step) {
        int lab = label[n];
        float s = X[n * HDIM + h] + 0.1f * X0[n * HDIM + h];
        atomicAdd(&sSum[lab * HDIM + h], s);
    }
    __syncthreads();
    for (int i = tid; i < NCLS * HDIM; i += 256) atomicAdd(&sums[i], sSum[i]);
}

// ---------------- centers + r2 = cm @ normalized pairwise diffs -------------
// grid 64 blocks x 128 threads; writes r2 transposed+padded f16 [128][64]
__global__ void k_centers(const float* __restrict__ sums, const float* __restrict__ cnt,
                          const float* __restrict__ cm, _Float16* __restrict__ r2t) {
    __shared__ float sCent[NCLS * HDIM];
    __shared__ float red[HDIM];
    int i = blockIdx.x, h = threadIdx.x;
    if (i >= NCLS) { r2t[h * 64 + i] = (_Float16)0.0f; return; }
    for (int idx = h; idx < NCLS * HDIM; idx += HDIM) {
        int cls = idx >> 7;
        sCent[idx] = sums[idx] / fmaxf(cnt[cls], 1.0f);
    }
    __syncthreads();
    float ci = sCent[i * HDIM + h];
    float acc = 0.0f;
    for (int j = 0; j < NCLS; ++j) {
        float d = ci - sCent[j * HDIM + h];
        red[h] = d * d;
        __syncthreads();
        for (int s = 64; s >= 1; s >>= 1) {
            if (h < s) red[h] += red[h + s];
            __syncthreads();
        }
        float rn = sqrtf(red[0]);
        rn = (rn == 0.0f) ? 1.0f : rn;
        acc += cm[i * NCLS + j] * d / rn;
        __syncthreads();
    }
    r2t[h * 64 + i] = (_Float16)acc;
}

// ======================= 128-K WMMA GEMM cores ==============================
// block: 256 thr (8 waves); 32 rows x 128 cols; wave -> col tile, 2 row tiles.

// lin0: out0 = out1 = relu(A@W + bias)
__global__ __launch_bounds__(256)
void k_gemm_lin0(const float* __restrict__ A, const float* __restrict__ W,
                 const float* __restrict__ bias,
                 float* __restrict__ out0, float* __restrict__ out1) {
    __shared__ _Float16 sA[32 * 136];
    __shared__ _Float16 sB[128 * 136];
    int tid = threadIdx.x;
    int wave = tid >> 5, lane = tid & 31;
    int half = lane >> 4, r16 = lane & 15;
    long rowBase = (long)blockIdx.x * 32;

    for (int idx = tid; idx < 32 * 128; idx += 256) {
        int rr = idx >> 7, cc = idx & 127;
        sA[rr * 136 + cc] = (_Float16)A[(rowBase + rr) * HDIM + cc];
    }
    for (int idx = tid; idx < 128 * 128; idx += 256) {   // W[k][n] -> sB[n][k]
        int kk = idx >> 7, nn = idx & 127;
        sB[nn * 136 + kk] = (_Float16)W[idx];
    }
    __syncthreads();

    int ct = wave;
    const _Float16* a0Base = &sA[r16 * 136];
    const _Float16* a1Base = &sA[(16 + r16) * 136];
    const _Float16* bBase  = &sB[(ct * 16 + r16) * 136];
    v8f c0 = {}, c1 = {};
#pragma unroll
    for (int kk = 0; kk < 4; ++kk) {
        v16h b  = load_frag(bBase, kk, half);
        v16h a0 = load_frag(a0Base, kk, half);
        v16h a1 = load_frag(a1Base, kk, half);
        c0 = __builtin_amdgcn_wmma_f32_16x16x32_f16(false, a0, false, b, (short)0, c0, false, false);
        c1 = __builtin_amdgcn_wmma_f32_16x16x32_f16(false, a1, false, b, (short)0, c1, false, false);
    }

    int col = ct * 16 + r16;
    float bb = bias[col];
#pragma unroll
    for (int v = 0; v < 8; ++v) {
        long row0 = rowBase + v + 8 * half;
        long row1 = row0 + 16;
        float v0 = fmaxf(c0[v] + bb, 0.0f);
        float v1 = fmaxf(c1[v] + bb, 0.0f);
        out0[row0 * HDIM + col] = v0;  out1[row0 * HDIM + col] = v0;
        out0[row1 * HDIM + col] = v1;  out1[row1 * HDIM + col] = v1;
    }
}

// conv: out = relu((1-beta)*U + beta*(U@W))   (A == U)
__global__ __launch_bounds__(256)
void k_gemm_conv(const float* __restrict__ U, const float* __restrict__ W,
                 float* __restrict__ out, float beta) {
    __shared__ _Float16 sA[32 * 136];
    __shared__ _Float16 sB[128 * 136];
    int tid = threadIdx.x;
    int wave = tid >> 5, lane = tid & 31;
    int half = lane >> 4, r16 = lane & 15;
    long rowBase = (long)blockIdx.x * 32;

    for (int idx = tid; idx < 32 * 128; idx += 256) {
        int rr = idx >> 7, cc = idx & 127;
        sA[rr * 136 + cc] = (_Float16)U[(rowBase + rr) * HDIM + cc];
    }
    for (int idx = tid; idx < 128 * 128; idx += 256) {
        int kk = idx >> 7, nn = idx & 127;
        sB[nn * 136 + kk] = (_Float16)W[idx];
    }
    __syncthreads();

    int ct = wave;
    const _Float16* a0Base = &sA[r16 * 136];
    const _Float16* a1Base = &sA[(16 + r16) * 136];
    const _Float16* bBase  = &sB[(ct * 16 + r16) * 136];
    v8f c0 = {}, c1 = {};
#pragma unroll
    for (int kk = 0; kk < 4; ++kk) {
        v16h b  = load_frag(bBase, kk, half);
        v16h a0 = load_frag(a0Base, kk, half);
        v16h a1 = load_frag(a1Base, kk, half);
        c0 = __builtin_amdgcn_wmma_f32_16x16x32_f16(false, a0, false, b, (short)0, c0, false, false);
        c1 = __builtin_amdgcn_wmma_f32_16x16x32_f16(false, a1, false, b, (short)0, c1, false, false);
    }

    int col = ct * 16 + r16;
    float omb = 1.0f - beta;
#pragma unroll
    for (int v = 0; v < 8; ++v) {
        long i0 = (rowBase + v + 8 * half) * HDIM + col;
        long i1 = i0 + 16 * HDIM;
        out[i0] = fmaxf(omb * U[i0] + beta * c0[v], 0.0f);
        out[i1] = fmaxf(omb * U[i1] + beta * c1[v], 0.0f);
    }
}

// ---------------- p @ r2 GEMM (K=47 padded to 64) + RSL combine -------------
// u = 0.45*(x + agg + p@r2) + 0.1*x0 ; written in-place into AGG.
// r2t (f16 [128][64], 16KB) staged via CDNA5 async global->LDS copy.
__global__ __launch_bounds__(256)
void k_pr_gemm(const float* __restrict__ P, const _Float16* __restrict__ R2T,
               const float* __restrict__ X, const float* __restrict__ X0,
               float* __restrict__ AGG) {
    __shared__ _Float16 sP[32 * 72];
    __shared__ _Float16 sR[128 * 72];
    int tid = threadIdx.x;
    int wave = tid >> 5, lane = tid & 31;
    int half = lane >> 4, r16 = lane & 15;
    long rowBase = (long)blockIdx.x * 32;

    // async-stage R2T rows (64 f16 = 128B each) into padded LDS (stride 72)
#pragma unroll
    for (int m = 0; m < 4; ++m) {
        int chunk = tid + 256 * m;              // 1024 chunks of 16B
        int nn = chunk >> 3;                    // row 0..127
        int c8 = chunk & 7;                     // 16B chunk within row
        unsigned lds_off = (unsigned)(size_t)&sR[nn * 72 + c8 * 8];
        unsigned goff    = (unsigned)((nn * 64 + c8 * 8) * 2);
        asm volatile("global_load_async_to_lds_b128 %0, %1, %2 offset:0"
                     :: "v"(lds_off), "v"(goff), "s"(R2T) : "memory");
    }

    for (int idx = tid; idx < 32 * 64; idx += 256) {
        int rr = idx >> 6, cc = idx & 63;
        float v = (cc < NCLS) ? P[(rowBase + rr) * NCLS + cc] : 0.0f;
        sP[rr * 72 + cc] = (_Float16)v;
    }
    asm volatile("s_wait_asynccnt 0" ::: "memory");
    __syncthreads();

    int ct = wave;
    const _Float16* a0Base = &sP[r16 * 72];
    const _Float16* a1Base = &sP[(16 + r16) * 72];
    const _Float16* bBase  = &sR[(ct * 16 + r16) * 72];
    v8f c0 = {}, c1 = {};
#pragma unroll
    for (int kk = 0; kk < 2; ++kk) {
        v16h b  = load_frag(bBase, kk, half);
        v16h a0 = load_frag(a0Base, kk, half);
        v16h a1 = load_frag(a1Base, kk, half);
        c0 = __builtin_amdgcn_wmma_f32_16x16x32_f16(false, a0, false, b, (short)0, c0, false, false);
        c1 = __builtin_amdgcn_wmma_f32_16x16x32_f16(false, a1, false, b, (short)0, c1, false, false);
    }

    int col = ct * 16 + r16;
#pragma unroll
    for (int v = 0; v < 8; ++v) {
        long i0 = (rowBase + v + 8 * half) * HDIM + col;
        long i1 = i0 + 16 * HDIM;
        AGG[i0] = 0.45f * (c0[v] + X[i0] + AGG[i0]) + 0.1f * X0[i0];
        AGG[i1] = 0.45f * (c1[v] + X[i1] + AGG[i1]) + 0.1f * X0[i1];
    }
}

// ---------------- final lin1: out[N,47] = X @ W1 + b1 -----------------------
// 96 threads (3 waves), 32 rows x 48 cols (col 47 guarded at store)
__global__ __launch_bounds__(96)
void k_lin1(const float* __restrict__ X, const float* __restrict__ W1,
            const float* __restrict__ b1, float* __restrict__ out) {
    __shared__ _Float16 sA[32 * 136];
    __shared__ _Float16 sB[48 * 136];
    int tid = threadIdx.x;
    int wave = tid >> 5, lane = tid & 31;
    int half = lane >> 4, r16 = lane & 15;
    long rowBase = (long)blockIdx.x * 32;

    for (int idx = tid; idx < 32 * 128; idx += 96) {
        int rr = idx >> 7, cc = idx & 127;
        sA[rr * 136 + cc] = (_Float16)X[(rowBase + rr) * HDIM + cc];
    }
    for (int idx = tid; idx < 48 * 128; idx += 96) {     // W1[k][n] -> sB[n][k], pad n=47
        int nn = idx >> 7, kk = idx & 127;
        float v = (nn < NCLS) ? W1[kk * NCLS + nn] : 0.0f;
        sB[nn * 136 + kk] = (_Float16)v;
    }
    __syncthreads();

    int ct = wave;                                       // 0..2
    const _Float16* a0Base = &sA[r16 * 136];
    const _Float16* a1Base = &sA[(16 + r16) * 136];
    const _Float16* bBase  = &sB[(ct * 16 + r16) * 136];
    v8f c0 = {}, c1 = {};
#pragma unroll
    for (int kk = 0; kk < 4; ++kk) {
        v16h b  = load_frag(bBase, kk, half);
        v16h a0 = load_frag(a0Base, kk, half);
        v16h a1 = load_frag(a1Base, kk, half);
        c0 = __builtin_amdgcn_wmma_f32_16x16x32_f16(false, a0, false, b, (short)0, c0, false, false);
        c1 = __builtin_amdgcn_wmma_f32_16x16x32_f16(false, a1, false, b, (short)0, c1, false, false);
    }

    int col = ct * 16 + r16;
    if (col < NCLS) {
        float bb = b1[col];
#pragma unroll
        for (int v = 0; v < 8; ++v) {
            long row0 = rowBase + v + 8 * half;
            long row1 = row0 + 16;
            out[row0 * NCLS + col] = c0[v] + bb;
            out[row1 * NCLS + col] = c1[v] + bb;
        }
    }
}

// ---------------------------------------------------------------------------
extern "C" void kernel_launch(void* const* d_in, const int* in_sizes, int n_in,
                              void* d_out, int out_size, void* d_ws, size_t ws_size,
                              hipStream_t stream) {
    const float* x      = (const float*)d_in[0];
    const int*   ei     = (const int*)  d_in[1];   // [2,E]
    const float* ew     = (const float*)d_in[2];
    const int*   label  = (const int*)  d_in[3];
    const float* p      = (const float*)d_in[4];
    const float* cm     = (const float*)d_in[5];
    const float* lin0_w = (const float*)d_in[6];
    const float* lin0_b = (const float*)d_in[7];
    const float* lin1_w = (const float*)d_in[8];
    const float* lin1_b = (const float*)d_in[9];
    const float* conv_w = (const float*)d_in[10];
    float* out = (float*)d_out;

    // workspace carve (~154 MB)
    char* ws = (char*)d_ws;
    float*    x0   = (float*)ws;            ws += (size_t)NH * 4;
    float*    xc   = (float*)ws;            ws += (size_t)NH * 4;
    float*    agg  = (float*)ws;            ws += (size_t)NH * 4;   // also holds u
    float*    sums = (float*)ws;            ws += (size_t)NCLS * HDIM * 4;
    float*    cnt  = (float*)ws;            ws += 256;
    _Float16* r2t  = (_Float16*)ws;         ws += 128 * 64 * 2;

    const int* src = ei;
    const int* dst = ei + E_EDGES;

    // per-class counts (constant across layers)
    k_zero<<<1, 64, 0, stream>>>(cnt, NCLS);
    k_count<<<(N_NODES + 255) / 256, 256, 0, stream>>>(label, cnt, N_NODES);

    // x = relu(x @ lin0_w + b0); x0 = x
    k_gemm_lin0<<<NBLK32, 256, 0, stream>>>(x, lin0_w, lin0_b, x0, xc);

    for (int i = 0; i < 4; ++i) {
        float beta = logf(0.5f / (float)(i + 1) + 1.0f);

        k_zero<<<4096, 256, 0, stream>>>(agg, (long)NH);
        k_scatter<<<E_EDGES / 8, 256, 0, stream>>>(src, dst, ew, xc, agg);

        k_zero<<<32, 256, 0, stream>>>(sums, (long)(NCLS * HDIM));
        k_classsum<<<512, 256, 0, stream>>>(xc, x0, label, sums);
        k_centers<<<64, 128, 0, stream>>>(sums, cnt, cm, r2t);

        // u = 0.45*(x + agg + p@r2) + 0.1*x0   (in-place into agg)
        k_pr_gemm<<<NBLK32, 256, 0, stream>>>(p, r2t, xc, x0, agg);

        // x = relu((1-beta)*u + beta*(u @ conv_w[i]))
        k_gemm_conv<<<NBLK32, 256, 0, stream>>>(agg, conv_w + (size_t)i * HDIM * HDIM,
                                                xc, beta);
    }

    // out = x @ lin1_w + b1
    k_lin1<<<NBLK32, 96, 0, stream>>>(xc, lin1_w, lin1_b, out);

    (void)in_sizes; (void)n_in; (void)out_size; (void)ws_size;
}
